// Mamba3_16423954940520
// MI455X (gfx1250) — matrix operand order repaired and checked by reading, not compiled, not run on hardware
//
#include <hip/hip_runtime.h>
#include <hip/hip_bf16.h>
#include <math.h>

#define LSEQ      1024
#define DMODEL    768
#define DSTATE    64
#define HEADDIM   64
#define DINNER    1536
#define NHEADS    24
#define SPLITN    32
#define N_ANG     16
#define DPROJ     3288
#define DPROJ_PAD 3296
#define EPSV      1e-5f
#define A_FLOOR   1e-4f

// proj column offsets
#define COL_Z     0
#define COL_X     1536
#define COL_B     3072
#define COL_C     3136
#define COL_DT    3200
#define COL_A     3224
#define COL_TRAP  3248
#define COL_ANG   3272

typedef __attribute__((ext_vector_type(16))) __bf16 v16bf;
typedef __attribute__((ext_vector_type(8)))  float  v8f;

static __device__ __forceinline__ unsigned short f32_to_bf16_rne(float f) {
  union { float f; unsigned int u; } v; v.f = f;
  unsigned int x = v.u;
  x += 0x7FFFu + ((x >> 16) & 1u);
  return (unsigned short)(x >> 16);
}
static __device__ __forceinline__ float softplus_f(float x) {
  return (x > 20.f) ? x : log1pf(expf(x));
}
static __device__ __forceinline__ float sigmoid_f(float x) {
  return 1.f / (1.f + expf(-x));
}

// ---------------- conversions ----------------
__global__ void k_cvt_bf16(const float* __restrict__ src,
                           unsigned short* __restrict__ dst, int n) {
  int i = blockIdx.x * blockDim.x + threadIdx.x;
  if (i < n) dst[i] = f32_to_bf16_rne(src[i]);
}

// W_in [3288x768] -> bf16 padded to 3296 rows (extra rows zero)
__global__ void k_cvt_win_pad(const float* __restrict__ src,
                              unsigned short* __restrict__ dst) {
  int i = blockIdx.x * blockDim.x + threadIdx.x;
  if (i >= DPROJ_PAD * DMODEL) return;
  int row = i / DMODEL;
  dst[i] = (row < DPROJ) ? f32_to_bf16_rne(src[i]) : (unsigned short)0;
}

// ---------------- WMMA GEMM: C[M,N] = A[M,K] @ B[N,K]^T (bf16 in, f32 out) ----
// Register-blocked 2x2 WMMA macro-tile: each wave owns a 32x32 output block.
// Each A fragment feeds 2 WMMAs and each B fragment feeds 2 WMMAs
// (16 flops/byte of operand traffic vs 8 for a 1x1 tile).
// 8 waves per 256-thread block. M%32==0, N%32==0, K%32==0 assumed.
__global__ void __launch_bounds__(256)
k_gemm_bf16(const unsigned short* __restrict__ A,
            const unsigned short* __restrict__ B,
            float* __restrict__ C,
            int M, int N, int K, int ldc) {
  const int warp = threadIdx.x >> 5;
  const int lane = threadIdx.x & 31;
  const int w = blockIdx.x * 8 + warp;
  const int ngrp = N >> 5;                 // 32-wide N groups
  const int mg = w / ngrp;
  const int ng = w - mg * ngrp;
  if (mg * 32 >= M) return;                // wave-uniform exit (EXEC stays all-1s)
  const int hi = lane >> 4;                // 0 or 1
  const int lo = lane & 15;

  const unsigned short* arow0 = A + (size_t)(mg * 32 + lo) * K;        // M rows 0..15
  const unsigned short* arow1 = arow0 + (size_t)16 * K;                // M rows 16..31
  const unsigned short* brow0 = B + (size_t)(ng * 32 + lo) * K;        // N cols 0..15
  const unsigned short* brow1 = brow0 + (size_t)16 * K;                // N cols 16..31

  v8f acc00 = {}, acc01 = {}, acc10 = {}, acc11 = {};
  for (int k0 = 0; k0 < K; k0 += 32) {
    union { uint4 q[2]; v16bf v; } a0, a1, b0, b1;
    // A 16x32 bf16: lane(lo,hi) holds K = {hi*8..hi*8+7} and {16+hi*8..}
    a0.q[0] = *(const uint4*)(arow0 + k0 + hi * 8);
    a0.q[1] = *(const uint4*)(arow0 + k0 + 16 + hi * 8);
    a1.q[0] = *(const uint4*)(arow1 + k0 + hi * 8);
    a1.q[1] = *(const uint4*)(arow1 + k0 + 16 + hi * 8);
    // B 32x16 bf16: lane(lo,hi) = column lo, K = {hi*8..hi*8+7} and {16+hi*8..}
    b0.q[0] = *(const uint4*)(brow0 + k0 + hi * 8);
    b0.q[1] = *(const uint4*)(brow0 + k0 + 16 + hi * 8);
    b1.q[0] = *(const uint4*)(brow1 + k0 + hi * 8);
    b1.q[1] = *(const uint4*)(brow1 + k0 + 16 + hi * 8);
    acc00 = __builtin_amdgcn_wmma_f32_16x16x32_bf16(false, a0.v, false, b0.v,
                                                    (short)0, acc00, false, false);
    acc01 = __builtin_amdgcn_wmma_f32_16x16x32_bf16(false, a0.v, false, b1.v,
                                                    (short)0, acc01, false, false);
    acc10 = __builtin_amdgcn_wmma_f32_16x16x32_bf16(false, a1.v, false, b0.v,
                                                    (short)0, acc10, false, false);
    acc11 = __builtin_amdgcn_wmma_f32_16x16x32_bf16(false, a1.v, false, b1.v,
                                                    (short)0, acc11, false, false);
  }
  // D layout: VGPR r -> M = r + 8*hi, N = lo (within each 16x16 sub-tile)
  {
    float* c00 = C + (size_t)(mg * 32 + hi * 8) * ldc + ng * 32 + lo;
    float* c10 = c00 + (size_t)16 * ldc;
#pragma unroll
    for (int r = 0; r < 8; ++r) {
      c00[(size_t)r * ldc]      = acc00[r];
      c00[(size_t)r * ldc + 16] = acc01[r];
      c10[(size_t)r * ldc]      = acc10[r];
      c10[(size_t)r * ldc + 16] = acc11[r];
    }
  }
}

// ---------------- per-timestep elementwise + RMSNorm ----------------
__global__ void __launch_bounds__(64)
k_ew1(const float* __restrict__ proj, const float* __restrict__ dt_bias,
      const float* __restrict__ Bnw, const float* __restrict__ Cnw,
      float* __restrict__ DT, float* __restrict__ aexp,
      float* __restrict__ trap, float* __restrict__ Bn, float* __restrict__ Cn) {
  const int l = blockIdx.x;
  const int t = threadIdx.x;
  const float* row = proj + (size_t)l * DPROJ_PAD;
  float b = row[COL_B + t];
  float c = row[COL_C + t];
  __shared__ float sb[64], sc[64];
  sb[t] = b * b; sc[t] = c * c;
  __syncthreads();
  for (int s = 32; s > 0; s >>= 1) {
    if (t < s) { sb[t] += sb[t + s]; sc[t] += sc[t + s]; }
    __syncthreads();
  }
  const float rb = sqrtf(sb[0] * (1.f / 64.f) + EPSV);
  const float rc = sqrtf(sc[0] * (1.f / 64.f) + EPSV);
  Bn[l * 64 + t] = b / rb * Bnw[t];
  Cn[l * 64 + t] = c / rc * Cnw[t];
  if (t < NHEADS) {
    const float dtv = softplus_f(row[COL_DT + t] + dt_bias[t]);
    const float Av  = -fmaxf(softplus_f(row[COL_A + t]), A_FLOOR);
    DT[l * NHEADS + t]   = dtv;
    aexp[l * NHEADS + t] = expf(Av * dtv);
    trap[l * NHEADS + t] = sigmoid_f(row[COL_TRAP + t]);
  }
}

// ---------------- angle cumsum: 384 independent prefix sums over L ----------
__global__ void __launch_bounds__(384)
k_angcum(const float* __restrict__ proj, const float* __restrict__ DT,
         float* __restrict__ ang) {
  const int t = threadIdx.x;          // 0..383
  const int h = t >> 4;
  const int j = t & 15;
  float acc = 0.f;
  for (int l = 0; l < LSEQ; ++l) {
    acc += proj[(size_t)l * DPROJ_PAD + COL_ANG + j] * DT[l * NHEADS + h];
    ang[((size_t)l * NHEADS + h) * N_ANG + j] = acc;
  }
}

// ---------------- bias + RoPE on B/C ----------------
__global__ void k_ropebias(const float* __restrict__ Bn, const float* __restrict__ Cn,
                           const float* __restrict__ ang,
                           const float* __restrict__ Bbias, const float* __restrict__ Cbias,
                           float* __restrict__ Bp, float* __restrict__ Cp) {
  int gid = blockIdx.x * blockDim.x + threadIdx.x;
  if (gid >= LSEQ * NHEADS * DSTATE) return;
  const int n  = gid & 63;
  const int lh = gid >> 6;
  const int h  = lh % NHEADS;
  const int l  = lh / NHEADS;
  float tb = Bn[l * 64 + n] + Bbias[h * 64 + n];
  float tc = Cn[l * 64 + n] + Cbias[h * 64 + n];
  if (n < SPLITN) {
    const int j = n >> 1;
    const float av = ang[((size_t)l * NHEADS + h) * N_ANG + j];
    const float cs = cosf(av), sn = sinf(av);
    const int pn = n ^ 1;
    const float ob = Bn[l * 64 + pn] + Bbias[h * 64 + pn];
    const float oc = Cn[l * 64 + pn] + Cbias[h * 64 + pn];
    if ((n & 1) == 0) { tb = tb * cs - ob * sn; tc = tc * cs - oc * sn; }
    else              { tb = ob * sn + tb * cs; tc = oc * sn + tc * cs; }
  }
  Bp[gid] = tb;
  Cp[gid] = tc;
}

// ---------------- sequential SSM scan: one block per head ----------------
// state h[64p x 64n] lives in registers: thread (p = tid/4, n-chunk = tid%4)
__global__ void __launch_bounds__(256)
k_scan(const float* __restrict__ proj, const float* __restrict__ aexp,
       const float* __restrict__ DT, const float* __restrict__ trap,
       const float* __restrict__ Bp, const float* __restrict__ Cp,
       const float* __restrict__ Dskip, float* __restrict__ yraw) {
  const int h = blockIdx.x;
  const int tid = threadIdx.x;
  const int p  = tid >> 2;
  const int nb = (tid & 3) << 4;
  __shared__ float xs[64], Bs[64], Cs[64];
  float hreg[16], bxp[16];
#pragma unroll
  for (int j = 0; j < 16; ++j) { hreg[j] = 0.f; bxp[j] = 0.f; }
  const float dsk = Dskip[h];

  for (int l = 0; l < LSEQ; ++l) {
    if (tid < 64)       xs[tid]       = proj[(size_t)l * DPROJ_PAD + COL_X + h * HEADDIM + tid];
    else if (tid < 128) Bs[tid - 64]  = Bp[((size_t)l * NHEADS + h) * DSTATE + (tid - 64)];
    else if (tid < 192) Cs[tid - 128] = Cp[((size_t)l * NHEADS + h) * DSTATE + (tid - 128)];
    const float a  = aexp[l * NHEADS + h];
    const float dt = DT[l * NHEADS + h];
    const float tr = trap[l * NHEADS + h];
    // pull next timestep's operands toward the WGP while this step computes
    if (l + 1 < LSEQ) {
      if (tid == 0)       __builtin_prefetch(&proj[(size_t)(l + 1) * DPROJ_PAD + COL_X + h * HEADDIM], 0, 0);
      else if (tid == 64) __builtin_prefetch(&Bp[((size_t)(l + 1) * NHEADS + h) * DSTATE], 0, 0);
      else if (tid == 128)__builtin_prefetch(&Cp[((size_t)(l + 1) * NHEADS + h) * DSTATE], 0, 0);
    }
    __syncthreads();
    const float x = xs[p];
    float part = 0.f;
#pragma unroll
    for (int j = 0; j < 16; ++j) {
      const float bx = x * Bs[nb + j];
      const float hv = a * hreg[j] + dt * ((1.f - tr) * bx + tr * a * bxp[j]);
      hreg[j] = hv;
      bxp[j] = bx;
      part += hv * Cs[nb + j];
    }
    // reduce the C-dot across the 4 lanes sharing row p (wave32 shuffle)
    part += __shfl_xor(part, 1, 32);
    part += __shfl_xor(part, 2, 32);
    if ((tid & 3) == 0)
      yraw[(size_t)l * DINNER + h * HEADDIM + p] = part + dsk * x;
    __syncthreads();
  }
}

// ---------------- gate y * silu(z), emit bf16 for GEMM2 ----------------
__global__ void k_gate(const float* __restrict__ yraw, const float* __restrict__ proj,
                       unsigned short* __restrict__ ybf) {
  int i = blockIdx.x * blockDim.x + threadIdx.x;
  if (i >= LSEQ * DINNER) return;
  const int l = i / DINNER;
  const int c = i - l * DINNER;
  const float z = proj[(size_t)l * DPROJ_PAD + COL_Z + c];
  const float v = yraw[i] * (z * sigmoid_f(z));
  ybf[i] = f32_to_bf16_rne(v);
}

// ---------------- host launch ----------------
extern "C" void kernel_launch(void* const* d_in, const int* in_sizes, int n_in,
                              void* d_out, int out_size, void* d_ws, size_t ws_size,
                              hipStream_t stream) {
  (void)in_sizes; (void)n_in; (void)out_size; (void)ws_size;
  const float* u        = (const float*)d_in[0];
  const float* W_in     = (const float*)d_in[1];
  const float* W_out    = (const float*)d_in[2];
  const float* dt_bias  = (const float*)d_in[3];
  const float* B_bias   = (const float*)d_in[4];
  const float* C_bias   = (const float*)d_in[5];
  const float* B_norm_w = (const float*)d_in[6];
  const float* C_norm_w = (const float*)d_in[7];
  const float* D_skip   = (const float*)d_in[8];
  float* out = (float*)d_out;

  char* ws = (char*)d_ws;
  size_t off = 0;
  auto alloc = [&](size_t bytes) -> void* {
    void* p = ws + off;
    off = (off + bytes + 255) & ~(size_t)255;
    return p;
  };

  float*          proj   = (float*)alloc((size_t)LSEQ * DPROJ_PAD * 4);
  unsigned short* ubf    = (unsigned short*)alloc((size_t)LSEQ * DMODEL * 2);
  unsigned short* winbf  = (unsigned short*)alloc((size_t)DPROJ_PAD * DMODEL * 2);
  unsigned short* woutbf = (unsigned short*)alloc((size_t)DMODEL * DINNER * 2);
  float*          DTb    = (float*)alloc((size_t)LSEQ * NHEADS * 4);
  float*          aexp   = (float*)alloc((size_t)LSEQ * NHEADS * 4);
  float*          trap   = (float*)alloc((size_t)LSEQ * NHEADS * 4);
  float*          Bn     = (float*)alloc((size_t)LSEQ * DSTATE * 4);
  float*          Cn     = (float*)alloc((size_t)LSEQ * DSTATE * 4);
  float*          ang    = (float*)alloc((size_t)LSEQ * NHEADS * N_ANG * 4);
  float*          Bp     = (float*)alloc((size_t)LSEQ * NHEADS * DSTATE * 4);
  float*          Cp     = (float*)alloc((size_t)LSEQ * NHEADS * DSTATE * 4);
  float*          yraw   = (float*)alloc((size_t)LSEQ * DINNER * 4);
  unsigned short* ybf    = (unsigned short*)alloc((size_t)LSEQ * DINNER * 2);

  // 1) bf16 conversions
  {
    int n = LSEQ * DMODEL;
    k_cvt_bf16<<<(n + 255) / 256, 256, 0, stream>>>(u, ubf, n);
  }
  {
    int n = DPROJ_PAD * DMODEL;
    k_cvt_win_pad<<<(n + 255) / 256, 256, 0, stream>>>(W_in, winbf);
  }
  {
    int n = DMODEL * DINNER;
    k_cvt_bf16<<<(n + 255) / 256, 256, 0, stream>>>(W_out, woutbf, n);
  }

  // 2) GEMM1: proj[1024,3296] = u[1024,768] @ W_in_pad[3296,768]^T
  //    32x32 per wave: (1024/32)*(3296/32) = 32*103 = 3296 waves = 412 blocks
  {
    int waves = (LSEQ / 32) * (DPROJ_PAD / 32);
    k_gemm_bf16<<<waves / 8, 256, 0, stream>>>(ubf, winbf, proj,
                                               LSEQ, DPROJ_PAD, DMODEL, DPROJ_PAD);
  }

  // 3) elementwise: DT / a=exp(A*DT) / trap / RMSNorm(B,C)
  k_ew1<<<LSEQ, 64, 0, stream>>>(proj, dt_bias, B_norm_w, C_norm_w,
                                 DTb, aexp, trap, Bn, Cn);

  // 4) angle cumsum over L (384 independent scans)
  k_angcum<<<1, 384, 0, stream>>>(proj, DTb, ang);

  // 5) bias + RoPE -> Bp, Cp [L, H, 64]
  {
    int n = LSEQ * NHEADS * DSTATE;
    k_ropebias<<<(n + 255) / 256, 256, 0, stream>>>(Bn, Cn, ang, B_bias, C_bias, Bp, Cp);
  }

  // 6) sequential SSM scan, one WGP-resident block per head
  k_scan<<<NHEADS, 256, 0, stream>>>(proj, aexp, DTb, trap, Bp, Cp, D_skip, yraw);

  // 7) gate y * silu(z) -> bf16
  {
    int n = LSEQ * DINNER;
    k_gate<<<(n + 255) / 256, 256, 0, stream>>>(yraw, proj, ybf);
  }

  // 8) GEMM2: out[1024,768] = ybf[1024,1536] @ W_out[768,1536]^T
  //    32x32 per wave: (1024/32)*(768/32) = 32*24 = 768 waves = 96 blocks
  {
    int waves = (LSEQ / 32) * (DMODEL / 32);
    k_gemm_bf16<<<waves / 8, 256, 0, stream>>>(ybf, woutbf, out,
                                               LSEQ, DMODEL, DINNER, DMODEL);
  }
}
